// StaticExpansionBlock_82162724372703
// MI455X (gfx1250) — matrix operand order repaired and checked by reading, not compiled, not run on hardware
//
// MI455X / gfx1250 implementation of StaticExpansionBlock.
//
// Roofline: ~292 GFLOP total, dominated by 4 projections of x (275 GFLOP).
// bf16 v_wmma_f32_16x16x32_bf16 (f32 accum) makes the GEMMs ~memory-bound
// (~1 GB HBM traffic @ 23.3 TB/s ~ 45us); fp32 WMMA (16x16x4) would be ~8x
// slower per instruction and compute-bound. All reductions / sigmoid / final
// blend stay in f32. Workspace requirement ~425 MB.
#include <hip/hip_runtime.h>
#include <hip/hip_bf16.h>
#include <stdint.h>

#define DM 1024
#define EX 64
#define NBATCH 16
#define LL 2048
#define EPSF 1e-9f

typedef __attribute__((ext_vector_type(16))) __bf16 bf16x16;
typedef __attribute__((ext_vector_type(8))) float f32x8;

union FragU {
  uint4 q[2];
  bf16x16 v;
};

static __device__ __forceinline__ uint32_t pack_bf16(float a, float b) {
  union { __bf16 h[2]; uint32_t u; } t;
  t.h[0] = (__bf16)a;
  t.h[1] = (__bf16)b;
  return t.u;
}

// LDS tiles hold packed (k,k+1) bf16 pairs: tile[row][kpair], 16 dwords/row (K_TILE=32).
// A-matrix lane layout (16-bit, 16x32): lane = 16*h + m; VGPR v holds
//   K = (v<4 ? 0 : 16) + 8*h + 2*(v&3)  -> v0..3 and v4..7 are each 4 consecutive dwords.
static __device__ __forceinline__ FragU frag_a(const uint32_t* l, int row, int h) {
  FragU f;
  f.q[0] = *(const uint4*)(l + row * 16 + 4 * h);
  f.q[1] = *(const uint4*)(l + row * 16 + 8 + 4 * h);
  return f;
}
// B-matrix lane layout (16-bit, 32x16): lane = 16*h + n; VGPR v holds K = (v<4 ? 0 : 16) + 8*h + 2*(v&3).
static __device__ __forceinline__ FragU frag_b(const uint32_t* l, int row, int h) {
  FragU f;
  f.q[0] = *(const uint4*)(l + row * 16 + 4 * h);
  f.q[1] = *(const uint4*)(l + row * 16 + 8 + 4 * h);
  return f;
}
static __device__ __forceinline__ f32x8 wmma_bf16(const FragU& a, const FragU& b, f32x8 c) {
  return __builtin_amdgcn_wmma_f32_16x16x32_bf16(false, a.v, false, b.v, (short)0, c, false, false);
}

// ---------------------------------------------------------------------------
// K1: f32 -> bf16 packs of x, weights, q_emb; zero the masked-sum accumulators.
// ---------------------------------------------------------------------------
__global__ __launch_bounds__(256) void k_prepack(
    const float* __restrict__ x, const float* __restrict__ Wk,
    const float* __restrict__ Wa, const float* __restrict__ Wb,
    const float* __restrict__ Ws, const float* __restrict__ qe,
    uint32_t* __restrict__ xbf, uint32_t* __restrict__ wbf,
    uint32_t* __restrict__ qbf, float* __restrict__ sae,
    float* __restrict__ sbe) {
  int i = blockIdx.x * 256 + threadIdx.x;
  const int N1 = NBATCH * LL * DM / 2;   // x pairs
  const int NW = DM * DM / 2;            // pairs per weight (2^19)
  const int N2 = N1 + 4 * NW;
  const int N3 = N2 + EX * DM / 2;
  const int N4 = N3 + NBATCH * EX;
  if (i < N1) {
    float2 v = ((const float2*)x)[i];
    xbf[i] = pack_bf16(v.x, v.y);
  } else if (i < N2) {
    int j = i - N1;
    int w = j >> 19, off = j & (NW - 1);
    const float* W = (w == 0) ? Wk : (w == 1) ? Wa : (w == 2) ? Wb : Ws;
    float2 v = ((const float2*)W)[off];
    wbf[j] = pack_bf16(v.x, v.y);
  } else if (i < N3) {
    int j = i - N2;
    float2 v = ((const float2*)qe)[j];
    qbf[j] = pack_bf16(v.x, v.y);
  } else if (i < N4) {
    int j = i - N3;
    sae[j] = 0.f;
    sbe[j] = 0.f;
  }
}

// ---------------------------------------------------------------------------
// K2: fused projection GEMM. C[r][c] = sum_k x[r][k] * W[c][k] + bias[c].
// grid = (M/128, D/128, 4 weights). 256 threads = 8 waves; each wave owns a
// 32x64 region (2 m-subtiles x 4 n-subtiles). Double-buffered LDS staging.
//   w=0 -> x_key  bf16 [B*L][D]
//   w=1 -> xa_t   bf16 [B][D][L]   (transposed for later B-operand use)
//   w=2 -> xb_t   bf16 [B][D][L]
//   w=3 -> sel    f32  [B*L][D] = sigmoid(.)
// ---------------------------------------------------------------------------
__global__ __launch_bounds__(256) void k_proj(
    const uint32_t* __restrict__ xbf, const uint32_t* __restrict__ wbf,
    const float* __restrict__ bk, const float* __restrict__ ba,
    const float* __restrict__ bb, const float* __restrict__ bs,
    __bf16* __restrict__ xkey, __bf16* __restrict__ xat,
    __bf16* __restrict__ xbt, float* __restrict__ sel) {
  __shared__ alignas(16) uint32_t lA[2][2048];
  __shared__ alignas(16) uint32_t lB[2][2048];
  const int mt = blockIdx.x, nt = blockIdx.y, w = blockIdx.z;
  const int tid = threadIdx.x;
  const int lane = tid & 31, wv = tid >> 5;
  const int h = lane >> 4, ln = lane & 15;
  const int wm = wv >> 1, wn = wv & 1;

  const f32x8 zero8 = {0.f, 0.f, 0.f, 0.f, 0.f, 0.f, 0.f, 0.f};
  f32x8 acc[2][4];
#pragma unroll
  for (int i = 0; i < 2; i++)
#pragma unroll
    for (int j = 0; j < 4; j++) acc[i][j] = zero8;

  const int lr = tid >> 1, lk = (tid & 1) * 8;  // 8 dwords per thread per tile
  const uint32_t* gA = xbf + ((size_t)(mt * 128 + lr)) * 512 + lk;
  const uint32_t* gB = wbf + (size_t)w * (DM * DM / 2) +
                       ((size_t)(nt * 128 + lr)) * 512 + lk;

  uint4 pa0, pa1, pb0, pb1;
  auto gload = [&](int ks) {
    const uint32_t* pA = gA + ks * 16;
    pa0 = *(const uint4*)pA;
    pa1 = *(const uint4*)(pA + 4);
    const uint32_t* pB = gB + ks * 16;
    pb0 = *(const uint4*)pB;
    pb1 = *(const uint4*)(pB + 4);
  };
  auto lstore = [&](int buf) {
    *(uint4*)(&lA[buf][lr * 16 + lk]) = pa0;
    *(uint4*)(&lA[buf][lr * 16 + lk + 4]) = pa1;
    *(uint4*)(&lB[buf][lr * 16 + lk]) = pb0;
    *(uint4*)(&lB[buf][lr * 16 + lk + 4]) = pb1;
  };

  gload(0);
  lstore(0);
  const int KS = DM / 32;
  for (int ks = 0; ks < KS; ks++) {
    __syncthreads();
    const int cur = ks & 1;
    if (ks + 1 < KS) gload(ks + 1);
    FragU fa[2], fb[4];
#pragma unroll
    for (int i = 0; i < 2; i++) fa[i] = frag_a(lA[cur], (2 * wm + i) * 16 + ln, h);
#pragma unroll
    for (int j = 0; j < 4; j++) fb[j] = frag_b(lB[cur], (4 * wn + j) * 16 + ln, h);
#pragma unroll
    for (int i = 0; i < 2; i++)
#pragma unroll
      for (int j = 0; j < 4; j++) acc[i][j] = wmma_bf16(fa[i], fb[j], acc[i][j]);
    if (ks + 1 < KS) lstore(cur ^ 1);
  }

  const float* bias = (w == 0) ? bk : (w == 1) ? ba : (w == 2) ? bb : bs;
#pragma unroll
  for (int i = 0; i < 2; i++) {
#pragma unroll
    for (int j = 0; j < 4; j++) {
      const int cg = nt * 128 + (4 * wn + j) * 16 + ln;
      const float bj = bias[cg];
#pragma unroll
      for (int v = 0; v < 8; v++) {
        const int rg = mt * 128 + (2 * wm + i) * 16 + v + 8 * h;
        const float val = acc[i][j][v] + bj;
        if (w == 0) {
          xkey[(size_t)rg * DM + cg] = (__bf16)val;
        } else if (w == 1) {
          const int b = rg >> 11, l = rg & (LL - 1);
          xat[((size_t)(b * DM + cg)) * LL + l] = (__bf16)val;
        } else if (w == 2) {
          const int b = rg >> 11, l = rg & (LL - 1);
          xbt[((size_t)(b * DM + cg)) * LL + l] = (__bf16)val;
        } else {
          sel[(size_t)rg * DM + cg] = 1.0f / (1.0f + __expf(-val));
        }
      }
    }
  }
}

// ---------------------------------------------------------------------------
// K3: z[b,e,l] = (q_emb[idx[b,e]] . x_key[b,l]) / 32, with fused reductions:
//   sa_e/sb_e[b,e] += mask * relu(+-z)  (global f32 atomics, 16 WGs/batch)
//   sa_l/sb_l[b,l]  = sum_e relu(+-z)   (exact per WG: full E in one tile)
// Tile 64(e) x 128(l), K=D=1024. grid = (L/128, B).
// ---------------------------------------------------------------------------
__global__ __launch_bounds__(256) void k_z(
    const uint32_t* __restrict__ qbf, const uint32_t* __restrict__ xkey,
    const int* __restrict__ nidx, const int* __restrict__ mask,
    float* __restrict__ zbuf, float* __restrict__ sae, float* __restrict__ sbe,
    float* __restrict__ sal, float* __restrict__ sbl) {
  __shared__ alignas(16) uint32_t lA[2][1024];
  __shared__ alignas(16) uint32_t lB[2][2048];
  __shared__ float colA[128], colB[128], rowA[64], rowB[64];
  const int lt = blockIdx.x, b = blockIdx.y;
  const int tid = threadIdx.x;
  const int lane = tid & 31, wv = tid >> 5;
  const int h = lane >> 4, ln = lane & 15;
  const int wm = wv >> 1, wn = wv & 1;

  if (tid < 128) { colA[tid] = 0.f; colB[tid] = 0.f; }
  if (tid < 64)  { rowA[tid] = 0.f; rowB[tid] = 0.f; }

  const f32x8 zero8 = {0.f, 0.f, 0.f, 0.f, 0.f, 0.f, 0.f, 0.f};
  f32x8 acc[4];
#pragma unroll
  for (int j = 0; j < 4; j++) acc[j] = zero8;

  const int ea = tid >> 2, ka = (tid & 3) * 4;  // A: 4 dwords/thread (gathered rows)
  const int rowidx = nidx[b * EX + ea];
  const uint32_t* gA = qbf + (size_t)rowidx * 512 + ka;
  const int nb = tid >> 1, kb = (tid & 1) * 8;  // B: 8 dwords/thread
  const uint32_t* gB = xkey + ((size_t)(b * LL + lt * 128 + nb)) * 512 + kb;

  uint4 pa, pb0, pb1;
  auto gload = [&](int ks) {
    pa = *(const uint4*)(gA + ks * 16);
    pb0 = *(const uint4*)(gB + ks * 16);
    pb1 = *(const uint4*)(gB + ks * 16 + 4);
  };
  auto lstore = [&](int buf) {
    *(uint4*)(&lA[buf][ea * 16 + ka]) = pa;
    *(uint4*)(&lB[buf][nb * 16 + kb]) = pb0;
    *(uint4*)(&lB[buf][nb * 16 + kb + 4]) = pb1;
  };

  gload(0);
  lstore(0);
  const int KS = DM / 32;
  for (int ks = 0; ks < KS; ks++) {
    __syncthreads();
    const int cur = ks & 1;
    if (ks + 1 < KS) gload(ks + 1);
    FragU fa = frag_a(lA[cur], wm * 16 + ln, h);
    FragU fb[4];
#pragma unroll
    for (int j = 0; j < 4; j++) fb[j] = frag_b(lB[cur], (4 * wn + j) * 16 + ln, h);
#pragma unroll
    for (int j = 0; j < 4; j++) acc[j] = wmma_bf16(fa, fb[j], acc[j]);
    if (ks + 1 < KS) lstore(cur ^ 1);
  }

#pragma unroll
  for (int j = 0; j < 4; j++) {
    const int cl = (4 * wn + j) * 16 + ln;
    const int lg = lt * 128 + cl;
    const float mf = (mask[b * LL + lg] != 0) ? 1.f : 0.f;
    float ca = 0.f, cb = 0.f;
#pragma unroll
    for (int v = 0; v < 8; v++) {
      const int eg = wm * 16 + v + 8 * h;
      const float zz = acc[j][v] * 0.03125f;  // 1/sqrt(1024)
      zbuf[((size_t)(b * EX + eg)) * LL + lg] = zz;
      const float za = fmaxf(zz, 0.f), zn = fmaxf(-zz, 0.f);
      ca += za;
      cb += zn;
      atomicAdd(&rowA[eg], za * mf);
      atomicAdd(&rowB[eg], zn * mf);
    }
    atomicAdd(&colA[cl], ca);
    atomicAdd(&colB[cl], cb);
  }
  __syncthreads();
  if (tid < 128) {
    sal[b * LL + lt * 128 + tid] = colA[tid];
    sbl[b * LL + lt * 128 + tid] = colB[tid];
  }
  if (tid < 64) {
    atomicAdd(&sae[b * EX + tid], rowA[tid]);
    atomicAdd(&sbe[b * EX + tid], rowB[tid]);
  }
}

// ---------------------------------------------------------------------------
// K4: class_{a,b}[b,e,d] = sum_l nfw[b,e,l] * (x@W.T+b)[b,l,d] + b_emb[idx],
// where nfw = mask*relu(+-z)/(sum+eps), built on the fly into bf16 A-tiles.
// Tile 64(e) x 128(d), K=L=2048. grid = (D/128, B, 2{a,b}).
// Output stored transposed: c{a,b}_t[b][d][e] f32.
// ---------------------------------------------------------------------------
__global__ __launch_bounds__(256) void k_class(
    const float* __restrict__ zbuf, const int* __restrict__ mask,
    const float* __restrict__ sae, const float* __restrict__ sbe,
    const uint32_t* __restrict__ xat, const uint32_t* __restrict__ xbt,
    const float* __restrict__ bemb, const int* __restrict__ nidx,
    float* __restrict__ cat_, float* __restrict__ cbt_) {
  __shared__ alignas(16) uint32_t lA[2][1024];
  __shared__ alignas(16) uint32_t lB[2][2048];
  __shared__ float sden[64];
  __shared__ int sidx[64];
  const int dt = blockIdx.x, b = blockIdx.y, ab = blockIdx.z;
  const float* sums = ab ? sbe : sae;
  const uint32_t* xt = ab ? xbt : xat;
  float* ct = ab ? cbt_ : cat_;
  const float sgn = ab ? -1.f : 1.f;
  const int tid = threadIdx.x;
  const int lane = tid & 31, wv = tid >> 5;
  const int h = lane >> 4, ln = lane & 15;
  const int wm = wv >> 1, wn = wv & 1;

  if (tid < 64) {
    sden[tid] = 1.f / (sums[b * EX + tid] + EPSF);
    sidx[tid] = nidx[b * EX + tid];
  }

  const f32x8 zero8 = {0.f, 0.f, 0.f, 0.f, 0.f, 0.f, 0.f, 0.f};
  f32x8 acc[4];
#pragma unroll
  for (int j = 0; j < 4; j++) acc[j] = zero8;

  const int ea = tid >> 2;          // e row of A-tile
  const int kpa = (tid & 3) * 4;    // kpair base (4 dwords = 8 l values)
  const float* gZ = zbuf + ((size_t)(b * EX + ea)) * LL + kpa * 2;
  const int* gM = mask + b * LL + kpa * 2;
  const int nb = tid >> 1, kb = (tid & 1) * 8;
  const uint32_t* gB = xt + ((size_t)(b * DM + dt * 128 + nb)) * (LL / 2) + kb;

  float4 pz0, pz1;
  int4 pm0, pm1;
  uint4 pb0, pb1;
  auto gload = [&](int ks) {
    pz0 = *(const float4*)(gZ + ks * 32);
    pz1 = *(const float4*)(gZ + ks * 32 + 4);
    pm0 = *(const int4*)(gM + ks * 32);
    pm1 = *(const int4*)(gM + ks * 32 + 4);
    pb0 = *(const uint4*)(gB + ks * 16);
    pb1 = *(const uint4*)(gB + ks * 16 + 4);
  };
  auto lstore = [&](int buf) {
    const float den = sden[ea];
    const float zf[8] = {pz0.x, pz0.y, pz0.z, pz0.w, pz1.x, pz1.y, pz1.z, pz1.w};
    const int mi[8] = {pm0.x, pm0.y, pm0.z, pm0.w, pm1.x, pm1.y, pm1.z, pm1.w};
#pragma unroll
    for (int p = 0; p < 4; p++) {
      const float w0 = (mi[2 * p]     != 0) ? fmaxf(sgn * zf[2 * p],     0.f) * den : 0.f;
      const float w1 = (mi[2 * p + 1] != 0) ? fmaxf(sgn * zf[2 * p + 1], 0.f) * den : 0.f;
      lA[buf][ea * 16 + kpa + p] = pack_bf16(w0, w1);
    }
    *(uint4*)(&lB[buf][nb * 16 + kb]) = pb0;
    *(uint4*)(&lB[buf][nb * 16 + kb + 4]) = pb1;
  };

  gload(0);
  __syncthreads();  // sden/sidx visible before first A-tile build
  lstore(0);
  const int KS = LL / 32;
  for (int ks = 0; ks < KS; ks++) {
    __syncthreads();
    const int cur = ks & 1;
    if (ks + 1 < KS) gload(ks + 1);
    FragU fa = frag_a(lA[cur], wm * 16 + ln, h);
    FragU fb[4];
#pragma unroll
    for (int j = 0; j < 4; j++) fb[j] = frag_b(lB[cur], (4 * wn + j) * 16 + ln, h);
#pragma unroll
    for (int j = 0; j < 4; j++) acc[j] = wmma_bf16(fa, fb[j], acc[j]);
    if (ks + 1 < KS) lstore(cur ^ 1);
  }

#pragma unroll
  for (int j = 0; j < 4; j++) {
    const int dg = dt * 128 + (4 * wn + j) * 16 + ln;
#pragma unroll
    for (int v = 0; v < 8; v++) {
      const int eg = wm * 16 + v + 8 * h;
      const float val = acc[j][v] + bemb[(size_t)sidx[eg] * DM + dg];
      ct[((size_t)(b * DM + dg)) * EX + eg] = val;
    }
  }
}

// ---------------------------------------------------------------------------
// K5: out[b,l,d] = sel*sum_e aw[b,l,e]*ca[b,e,d] + (1-sel)*sum_e bw*cb, with
// aw/bw = relu(+-z^T)/(colsum+eps) built into bf16 A-tiles. Tile 64(l)x128(d),
// K=E=64 (2 steps). grid = (L/64, D/128, B).
// ---------------------------------------------------------------------------
__global__ __launch_bounds__(256) void k_final(
    const float* __restrict__ zbuf, const float* __restrict__ sal,
    const float* __restrict__ sbl, const float* __restrict__ cat_,
    const float* __restrict__ cbt_, const float* __restrict__ sel,
    float* __restrict__ out) {
  __shared__ alignas(16) uint32_t lAa[1024], lAb[1024];
  __shared__ alignas(16) uint32_t lBa[2048], lBb[2048];
  __shared__ float sdA[64], sdB[64];
  const int lt = blockIdx.x, dt = blockIdx.y, b = blockIdx.z;
  const int tid = threadIdx.x;
  const int lane = tid & 31, wv = tid >> 5;
  const int h = lane >> 4, ln = lane & 15;
  const int wm = wv >> 1, wn = wv & 1;

  if (tid < 64) {
    sdA[tid] = 1.f / (sal[b * LL + lt * 64 + tid] + EPSF);
    sdB[tid] = 1.f / (sbl[b * LL + lt * 64 + tid] + EPSF);
  }

  const f32x8 zero8 = {0.f, 0.f, 0.f, 0.f, 0.f, 0.f, 0.f, 0.f};
  f32x8 accA[4], accB[4];
#pragma unroll
  for (int j = 0; j < 4; j++) { accA[j] = zero8; accB[j] = zero8; }

  for (int ks = 0; ks < 2; ks++) {
    __syncthreads();  // also covers sdA/sdB fill on ks==0
    // A-tiles: z^T with relu + per-l normalization (both signs from one read)
    for (int t = tid; t < 1024; t += 256) {
      const int m = t & 63, kp = t >> 6;
      const int e0 = ks * 32 + 2 * kp;
      const int lg = lt * 64 + m;
      const float z0 = zbuf[((size_t)(b * EX + e0)) * LL + lg];
      const float z1 = zbuf[((size_t)(b * EX + e0 + 1)) * LL + lg];
      lAa[m * 16 + kp] = pack_bf16(fmaxf(z0, 0.f) * sdA[m], fmaxf(z1, 0.f) * sdA[m]);
      lAb[m * 16 + kp] = pack_bf16(fmaxf(-z0, 0.f) * sdB[m], fmaxf(-z1, 0.f) * sdB[m]);
    }
    // B-tiles: class_{a,b}^T rows (k=e contiguous f32 -> packed bf16 pairs)
    for (int t = tid; t < 1024; t += 256) {
      const int n = t >> 3, kpp = t & 7;
      const size_t base = ((size_t)(b * DM + dt * 128 + n)) * EX + ks * 32 + kpp * 4;
      const float4 va = *(const float4*)(cat_ + base);
      const float4 vb = *(const float4*)(cbt_ + base);
      lBa[n * 16 + kpp * 2] = pack_bf16(va.x, va.y);
      lBa[n * 16 + kpp * 2 + 1] = pack_bf16(va.z, va.w);
      lBb[n * 16 + kpp * 2] = pack_bf16(vb.x, vb.y);
      lBb[n * 16 + kpp * 2 + 1] = pack_bf16(vb.z, vb.w);
    }
    __syncthreads();
    const FragU faA = frag_a(lAa, wm * 16 + ln, h);
    const FragU faB = frag_a(lAb, wm * 16 + ln, h);
#pragma unroll
    for (int j = 0; j < 4; j++) {
      const FragU fbA = frag_b(lBa, (4 * wn + j) * 16 + ln, h);
      const FragU fbB = frag_b(lBb, (4 * wn + j) * 16 + ln, h);
      accA[j] = wmma_bf16(faA, fbA, accA[j]);
      accB[j] = wmma_bf16(faB, fbB, accB[j]);
    }
  }

#pragma unroll
  for (int j = 0; j < 4; j++) {
    const int dg = dt * 128 + (4 * wn + j) * 16 + ln;
#pragma unroll
    for (int v = 0; v < 8; v++) {
      const int lg = lt * 64 + wm * 16 + v + 8 * h;
      const size_t oi = ((size_t)(b * LL + lg)) * DM + dg;
      const float s = sel[oi];
      out[oi] = s * accA[j][v] + (1.f - s) * accB[j][v];
    }
  }
}

// ---------------------------------------------------------------------------
extern "C" void kernel_launch(void* const* d_in, const int* in_sizes, int n_in,
                              void* d_out, int out_size, void* d_ws, size_t ws_size,
                              hipStream_t stream) {
  (void)in_sizes; (void)n_in; (void)out_size; (void)ws_size;
  const float* x  = (const float*)d_in[0];
  const int* nidx = (const int*)d_in[1];
  const int* mask = (const int*)d_in[2];
  const float* qe = (const float*)d_in[3];
  const float* be = (const float*)d_in[4];
  const float* Wk = (const float*)d_in[5];
  const float* bk = (const float*)d_in[6];
  const float* Wa = (const float*)d_in[7];
  const float* ba = (const float*)d_in[8];
  const float* Wb = (const float*)d_in[9];
  const float* bb = (const float*)d_in[10];
  const float* Ws = (const float*)d_in[11];
  const float* bs = (const float*)d_in[12];
  float* out = (float*)d_out;

  // Workspace carve-out (~425 MB total), 256B aligned.
  char* w = (char*)d_ws;
  size_t o = 0;
  auto carve = [&](size_t bytes) -> void* {
    void* p = w + o;
    o = (o + bytes + 255) & ~(size_t)255;
    return p;
  };
  uint32_t* xbf = (uint32_t*)carve((size_t)NBATCH * LL * DM * 2);
  uint32_t* wbf = (uint32_t*)carve((size_t)4 * DM * DM * 2);
  uint32_t* qbf = (uint32_t*)carve((size_t)EX * DM * 2);
  void* xkey = carve((size_t)NBATCH * LL * DM * 2);
  void* xat  = carve((size_t)NBATCH * DM * LL * 2);
  void* xbt  = carve((size_t)NBATCH * DM * LL * 2);
  float* sel  = (float*)carve((size_t)NBATCH * LL * DM * 4);
  float* zbuf = (float*)carve((size_t)NBATCH * EX * LL * 4);
  float* sae  = (float*)carve((size_t)NBATCH * EX * 4);
  float* sbe  = (float*)carve((size_t)NBATCH * EX * 4);
  float* sal  = (float*)carve((size_t)NBATCH * LL * 4);
  float* sbl  = (float*)carve((size_t)NBATCH * LL * 4);
  float* cat_ = (float*)carve((size_t)NBATCH * DM * EX * 4);
  float* cbt_ = (float*)carve((size_t)NBATCH * DM * EX * 4);

  const int total = NBATCH * LL * DM / 2 + 4 * DM * DM / 2 + EX * DM / 2 + NBATCH * EX;
  k_prepack<<<(total + 255) / 256, 256, 0, stream>>>(x, Wk, Wa, Wb, Ws, qe,
                                                     xbf, wbf, qbf, sae, sbe);
  k_proj<<<dim3(NBATCH * LL / 128, DM / 128, 4), 256, 0, stream>>>(
      xbf, wbf, bk, ba, bb, bs, (__bf16*)xkey, (__bf16*)xat, (__bf16*)xbt, sel);
  k_z<<<dim3(LL / 128, NBATCH), 256, 0, stream>>>(
      qbf, (const uint32_t*)xkey, nidx, mask, zbuf, sae, sbe, sal, sbl);
  k_class<<<dim3(DM / 128, NBATCH, 2), 256, 0, stream>>>(
      zbuf, mask, sae, sbe, (const uint32_t*)xat, (const uint32_t*)xbt, be, nidx,
      cat_, cbt_);
  k_final<<<dim3(LL / 64, DM / 128, NBATCH), 256, 0, stream>>>(
      zbuf, sal, sbl, cat_, cbt_, sel, out);
}
